// BI_LSTM_Model_44049184587871
// MI455X (gfx1250) — compile-verified, matrix operand order checked by reading, not run on hardware
//
#include <hip/hip_runtime.h>
#include <hip/hip_bf16.h>

// ---------------------------------------------------------------------------
// Bi-LSTM (3 layers, projected, fw+bw) for MI455X / gfx1250.
// GEMMs use V_WMMA_F32_16X16X32_BF16 (bf16 operands, fp32 accumulate).
// Cell state (fp32) and biases live in LDS; h feeds back through the LDS
// A-operand buffer. Weights pre-packed into fully-coalesced bf16 B-fragments.
// Kernel is templated on layer shape so the K-loop fully unrolls and all
// fragment loads become base + immediate-offset (no per-load address math).
// ---------------------------------------------------------------------------

typedef __attribute__((ext_vector_type(16))) __bf16 v16bf;
typedef __attribute__((ext_vector_type(8)))  float  v8f;

#define T_STEPS 160
#define B_FULL  640
#define HID     768
#define PROJ    256
#define NGATE   3072   // 4*HID

__device__ __forceinline__ unsigned short f2bf(float f) {
  unsigned int u = __float_as_uint(f);
  unsigned int r = u + 0x7FFFu + ((u >> 16) & 1u);   // round-to-nearest-even
  return (unsigned short)(r >> 16);
}
__device__ __forceinline__ float bf2f(unsigned short b) {
  return __uint_as_float(((unsigned int)b) << 16);
}
__device__ __forceinline__ float sigmoidf(float x) {
  return 1.0f / (1.0f + __expf(-x));
}
__device__ __forceinline__ float tanh_fast(float x) {
  return 2.0f / (1.0f + __expf(-2.0f * x)) - 1.0f;
}

// ---------------------------------------------------------------------------
// Pack a row-major fp32 weight [Krows x N] into bf16 WMMA B-fragments.
// Packed layout (per 32x16 K-N tile, 512 bf16): [half h][lane][8 elems], i.e.
//   element gid = ((tile*2 + h)*32 + lane)*8 + e2,  tile = nt*KT + kt
//   k = kt*32 + (lane>>4)*8 + h*16 + e2,  n = nt*16 + (lane&15)
// => in the GEMM each lane loads 16 contiguous bytes and a whole
//    global_load_b128 covers 512 contiguous bytes (perfect coalescing).
// Rows k >= Krows (K padding) are zero-filled.
// ---------------------------------------------------------------------------
__global__ void pack_wmma_b(const float* __restrict__ src,
                            unsigned short* __restrict__ dst,
                            int Krows, int Kpad, int N) {
  const int KT = Kpad >> 5;
  const size_t total = (size_t)Kpad * (size_t)N;
  for (size_t gid = (size_t)blockIdx.x * blockDim.x + threadIdx.x; gid < total;
       gid += (size_t)gridDim.x * blockDim.x) {
    const int e2   = (int)(gid & 7);
    const int lane = (int)((gid >> 3) & 31);
    const int h    = (int)((gid >> 8) & 1);
    const size_t tile = gid >> 9;
    const int kt = (int)(tile % (size_t)KT);
    const int nt = (int)(tile / (size_t)KT);
    const int k = kt * 32 + ((lane >> 4) << 3) + (h << 4) + e2;
    const int n = nt * 16 + (lane & 15);
    const float v = (k < Krows) ? src[(size_t)k * N + n] : 0.0f;
    dst[gid] = f2bf(v);
  }
}

// ---------------------------------------------------------------------------
// One LSTM-with-projection layer, full T-step scan.
// Grid: 40 workgroups (B/16 M-tiles) x 256 threads (8 waves).
// Each wave owns 6 cell-column tiles (48x16 = HID) and 2 projection n-tiles.
// ---------------------------------------------------------------------------
template <int IN_DIM, int KPAD, bool XF32>
__global__ __launch_bounds__(256)
void lstm_layer(const void* __restrict__ xin,
                const unsigned short* __restrict__ wk,   // packed [4*48][KT] tiles
                const float* __restrict__ bias,          // [4*HID], order i,j,f,o
                const unsigned short* __restrict__ wp,   // packed [16][24] tiles
                unsigned short* __restrict__ yout,       // [T][B][PROJ] bf16 | null
                float* __restrict__ ep,                  // [2][B][PROJ] fp32 | null
                int reverse) {
  constexpr int KT = KPAD / 32;
  constexpr int AK = KPAD + 8;    // padded LDS row strides (bank rotation)
  constexpr int SK = HID + 8;
  constexpr int CK = HID + 4;

  __shared__ __align__(32) unsigned short a_buf[16 * AK];  // A operand: [xt | h | pad]
  __shared__ __align__(32) unsigned short s_buf[16 * SK];  // s = sig(o)*tanh(c)
  __shared__ __align__(16) float          c_lds[16 * CK];  // fp32 cell state
  __shared__ __align__(16) float          b_lds[NGATE];    // biases (+1 on f)

  const int tid   = threadIdx.x;
  const int wave  = tid >> 5;
  const int lane  = tid & 31;
  const int l15   = lane & 15;   // A: row m; B/C: column n
  const int khalf = lane >> 4;   // A/B: K-half select; C: row-half select
  const int row0  = blockIdx.x * 16;

  // One-time init: zero h+pad region of A, zero cell state, stage biases.
  for (int idx = tid; idx < 16 * (KPAD - IN_DIM); idx += 256) {
    const int m = idx / (KPAD - IN_DIM);
    const int j = idx - m * (KPAD - IN_DIM);
    a_buf[m * AK + IN_DIM + j] = 0;
  }
  for (int idx = tid; idx < 16 * CK; idx += 256) c_lds[idx] = 0.0f;
  for (int idx = tid; idx < NGATE; idx += 256) {
    float v = bias[idx];
    if (idx >= 2 * HID && idx < 3 * HID) v += 1.0f;   // forget_bias = 1.0
    b_lds[idx] = v;
  }

  const float*          xf = (const float*)xin;
  const unsigned short* xb = (const unsigned short*)xin;

  for (int step = 0; step < T_STEPS; ++step) {
    const int t = reverse ? (T_STEPS - 1 - step) : step;

    // ---- Phase 1: stage x_t tile into A buffer (bf16) ----------------------
    for (int idx = tid; idx < 16 * IN_DIM; idx += 256) {
      const int m = idx / IN_DIM;
      const int k = idx - m * IN_DIM;
      const size_t src = ((size_t)t * B_FULL + row0 + m) * (size_t)IN_DIM + k;
      a_buf[m * AK + k] = XF32 ? f2bf(xf[src]) : xb[src];
    }
    __syncthreads();

    // ---- Phase 2: z = A @ Wk, fused i/j/f/o gate math ----------------------
    const unsigned short* const pA = a_buf + l15 * AK + khalf * 8;
#pragma unroll 1
    for (int ct0 = 0; ct0 < 6; ++ct0) {
      const int ct = wave * 6 + ct0;
      const unsigned short* const pB0 = wk + (size_t)((0 * 48 + ct) * KT) * 512 + lane * 8;
      const unsigned short* const pB1 = wk + (size_t)((1 * 48 + ct) * KT) * 512 + lane * 8;
      const unsigned short* const pB2 = wk + (size_t)((2 * 48 + ct) * KT) * 512 + lane * 8;
      const unsigned short* const pB3 = wk + (size_t)((3 * 48 + ct) * KT) * 512 + lane * 8;
      v8f acc0 = {}, acc1 = {}, acc2 = {}, acc3 = {};
#pragma unroll
      for (int kt = 0; kt < KT; ++kt) {
        union { v16bf v; uint4 q[2]; } a, b0, b1, b2, b3;
        a.q[0]  = *(const uint4*)(pA + kt * 32);
        a.q[1]  = *(const uint4*)(pA + kt * 32 + 16);
        b0.q[0] = *(const uint4*)(pB0 + kt * 512);
        b0.q[1] = *(const uint4*)(pB0 + kt * 512 + 256);
        b1.q[0] = *(const uint4*)(pB1 + kt * 512);
        b1.q[1] = *(const uint4*)(pB1 + kt * 512 + 256);
        b2.q[0] = *(const uint4*)(pB2 + kt * 512);
        b2.q[1] = *(const uint4*)(pB2 + kt * 512 + 256);
        b3.q[0] = *(const uint4*)(pB3 + kt * 512);
        b3.q[1] = *(const uint4*)(pB3 + kt * 512 + 256);
        acc0 = __builtin_amdgcn_wmma_f32_16x16x32_bf16(false, a.v, false, b0.v, (short)0, acc0, false, false);
        acc1 = __builtin_amdgcn_wmma_f32_16x16x32_bf16(false, a.v, false, b1.v, (short)0, acc1, false, false);
        acc2 = __builtin_amdgcn_wmma_f32_16x16x32_bf16(false, a.v, false, b2.v, (short)0, acc2, false, false);
        acc3 = __builtin_amdgcn_wmma_f32_16x16x32_bf16(false, a.v, false, b3.v, (short)0, acc3, false, false);
      }
      // Gate math; C-frag element (m = r + khalf*8, n = ct*16 + l15).
      const int   nc = ct * 16 + l15;
      const float bi = b_lds[0 * HID + nc];
      const float bj = b_lds[1 * HID + nc];
      const float bf = b_lds[2 * HID + nc];   // includes +1
      const float bo = b_lds[3 * HID + nc];
#pragma unroll
      for (int r = 0; r < 8; ++r) {
        const int m = r + khalf * 8;
        const float cprev = c_lds[m * CK + nc];
        const float cn = sigmoidf(acc2[r] + bf) * cprev +
                         sigmoidf(acc0[r] + bi) * tanh_fast(acc1[r] + bj);
        c_lds[m * CK + nc] = cn;
        s_buf[m * SK + nc] = f2bf(sigmoidf(acc3[r] + bo) * tanh_fast(cn));
      }
    }
    __syncthreads();

    // ---- Phase 3: h = s @ Wp; feed next step + emit outputs ----------------
    const unsigned short* const pS = s_buf + l15 * SK + khalf * 8;
#pragma unroll 1
    for (int pp = 0; pp < 2; ++pp) {
      const int nt = wave * 2 + pp;
      const unsigned short* const pW = wp + (size_t)(nt * (HID / 32)) * 512 + lane * 8;
      v8f hacc = {};
#pragma unroll
      for (int kt = 0; kt < HID / 32; ++kt) {
        union { v16bf v; uint4 q[2]; } a, b;
        a.q[0] = *(const uint4*)(pS + kt * 32);
        a.q[1] = *(const uint4*)(pS + kt * 32 + 16);
        b.q[0] = *(const uint4*)(pW + kt * 512);
        b.q[1] = *(const uint4*)(pW + kt * 512 + 256);
        hacc = __builtin_amdgcn_wmma_f32_16x16x32_bf16(false, a.v, false, b.v, (short)0, hacc, false, false);
      }
#pragma unroll
      for (int r = 0; r < 8; ++r) {
        const int m = r + khalf * 8;
        const int u = nt * 16 + l15;
        const unsigned short hb = f2bf(hacc[r]);
        a_buf[m * AK + IN_DIM + u] = hb;  // h operand for next step
        if (yout) yout[((size_t)t * B_FULL + row0 + m) * PROJ + u] = hb;
        if (ep && (t == 0 || t == T_STEPS - 1))
          ep[((size_t)(t == 0 ? 0 : 1) * B_FULL + row0 + m) * PROJ + u] = bf2f(hb);
      }
    }
    __syncthreads();
  }
}

// ---------------------------------------------------------------------------
// embedded = l2_normalize((outputs[0] + outputs[T-1]) / 2), outputs = [fw|bw].
// ---------------------------------------------------------------------------
__global__ void embed_norm(const float* __restrict__ ep_fw,
                           const float* __restrict__ ep_bw,
                           float* __restrict__ out) {
  const int row = blockIdx.x;
  const int tid = threadIdx.x;          // 256 threads, 2 columns each
  __shared__ float red[256];
  float vals[2];
  float ss = 0.0f;
#pragma unroll
  for (int j = 0; j < 2; ++j) {
    const int col = tid * 2 + j;        // 0..511
    const float* src = (col < PROJ) ? ep_fw : ep_bw;
    const int u = col & (PROJ - 1);
    const float a = src[(size_t)row * PROJ + u];                       // t = 0
    const float b = src[((size_t)B_FULL + row) * PROJ + u];            // t = T-1
    const float e = 0.5f * (a + b);
    vals[j] = e;
    ss += e * e;
  }
  red[tid] = ss;
  __syncthreads();
  for (int s = 128; s > 0; s >>= 1) {
    if (tid < s) red[tid] += red[tid + s];
    __syncthreads();
  }
  const float inv = rsqrtf(fmaxf(red[0], 1e-12f));
#pragma unroll
  for (int j = 0; j < 2; ++j) out[(size_t)row * 512 + tid * 2 + j] = vals[j] * inv;
}

// ---------------------------------------------------------------------------
extern "C" void kernel_launch(void* const* d_in, const int* in_sizes, int n_in,
                              void* d_out, int out_size, void* d_ws, size_t ws_size,
                              hipStream_t stream) {
  (void)in_sizes; (void)n_in; (void)out_size; (void)ws_size;

  const float* x = (const float*)d_in[0];
  const float* Wk[6]; const float* Bb[6]; const float* Wp[6];
  for (int i = 0; i < 6; ++i) {          // 0..2 = fw layers, 3..5 = bw layers
    Wk[i] = (const float*)d_in[1 + i * 3 + 0];
    Bb[i] = (const float*)d_in[1 + i * 3 + 1];
    Wp[i] = (const float*)d_in[1 + i * 3 + 2];
  }
  const int Kpad_l[3]  = {320, 512, 512};   // (40|256)+256 rounded up to 32
  const int Krows_l[3] = {296, 512, 512};

  char* ws = (char*)d_ws;
  size_t off = 0;
  auto alloc = [&](size_t bytes) -> void* {
    void* p = ws + off;
    off += (bytes + 255) & ~(size_t)255;
    return p;
  };

  unsigned short* wkp[6]; unsigned short* wpp[6];
  for (int i = 0; i < 6; ++i) {
    const int l = i % 3;
    wkp[i] = (unsigned short*)alloc((size_t)Kpad_l[l] * NGATE * sizeof(unsigned short));
    wpp[i] = (unsigned short*)alloc((size_t)HID * PROJ * sizeof(unsigned short));
  }
  unsigned short* bufA = (unsigned short*)alloc((size_t)T_STEPS * B_FULL * PROJ * sizeof(unsigned short));
  unsigned short* bufB = (unsigned short*)alloc((size_t)T_STEPS * B_FULL * PROJ * sizeof(unsigned short));
  float* ep_fw = (float*)alloc((size_t)2 * B_FULL * PROJ * sizeof(float));
  float* ep_bw = (float*)alloc((size_t)2 * B_FULL * PROJ * sizeof(float));

  // Pack all weights to bf16 WMMA fragments (re-done every call: deterministic).
  for (int i = 0; i < 6; ++i) {
    const int l = i % 3;
    const size_t totK = (size_t)Kpad_l[l] * NGATE;
    pack_wmma_b<<<(int)((totK + 255) / 256), 256, 0, stream>>>(Wk[i], wkp[i], Krows_l[l], Kpad_l[l], NGATE);
    pack_wmma_b<<<(HID * PROJ + 255) / 256, 256, 0, stream>>>(Wp[i], wpp[i], HID, HID, PROJ);
  }

  // Forward stack.
  lstm_layer<40, 320, true ><<<B_FULL / 16, 256, 0, stream>>>(x,    wkp[0], Bb[0], wpp[0], bufA,    nullptr, 0);
  lstm_layer<256, 512, false><<<B_FULL / 16, 256, 0, stream>>>(bufA, wkp[1], Bb[1], wpp[1], bufB,    nullptr, 0);
  lstm_layer<256, 512, false><<<B_FULL / 16, 256, 0, stream>>>(bufB, wkp[2], Bb[2], wpp[2], nullptr, ep_fw,  0);
  // Backward stack (buffers reused: stream serializes).
  lstm_layer<40, 320, true ><<<B_FULL / 16, 256, 0, stream>>>(x,    wkp[3], Bb[3], wpp[3], bufA,    nullptr, 1);
  lstm_layer<256, 512, false><<<B_FULL / 16, 256, 0, stream>>>(bufA, wkp[4], Bb[4], wpp[4], bufB,    nullptr, 1);
  lstm_layer<256, 512, false><<<B_FULL / 16, 256, 0, stream>>>(bufB, wkp[5], Bb[5], wpp[5], nullptr, ep_bw,  1);

  embed_norm<<<B_FULL, 256, 0, stream>>>(ep_fw, ep_bw, (float*)d_out);
}
